// AGRNN_77232101917046
// MI455X (gfx1250) — matrix operand order repaired and checked by reading, not compiled, not run on hardware
//
#include <hip/hip_runtime.h>
#include <math.h>

typedef _Float16 v16h __attribute__((ext_vector_type(16)));
typedef _Float16 v8h  __attribute__((ext_vector_type(8)));
typedef float    v8f  __attribute__((ext_vector_type(8)));

#define NN    20000
#define EE    320000
#define DD    512
#define DWW   300
#define DSS   16
#define NC    13

#define KP_E  1056   // [src 512 | s_f 16 | dst 512 | pad 16]  = 33*32
#define KP_L  608    // [src 304 | dst 304]                    = 19*32
#define KP_N  1024
#define KP_NL 608
#define KP_R  1664   // [nsrc 512 | wsrc 304 | s_f 16 | wdst 304 | ndst 512 | pad 16]
#define DWP   304    // padded word2vec width (19*16)

// ---------------- CDNA5 async global->LDS copy (ASYNCcnt) ------------------

__device__ __forceinline__ void async_cp16(void* lds, const void* g) {
  unsigned l = (unsigned)(size_t)lds;                 // LDS byte offset
  unsigned long long ga = (unsigned long long)(size_t)g;
  asm volatile("global_load_async_to_lds_b128 %0, %1, off"
               :: "v"(l), "v"(ga) : "memory");
}
__device__ __forceinline__ void async_wait0() {
  asm volatile("s_wait_asynccnt 0" ::: "memory");
}

// ---------------- WMMA helpers (wave32, 16x16x32 f16) ----------------------

__device__ __forceinline__ v8f wmma16(v16h a, v16h b, v8f c) {
  return __builtin_amdgcn_wmma_f32_16x16x32_f16(false, a, false, b, (short)0, c,
                                                false, false);
}

// A fragment from row-major [16 x stride] f16 LDS tile.
__device__ __forceinline__ v16h load_a_lds(const _Float16* base, int stride,
                                           int lane, int k0) {
  const int m  = lane & 15;
  const int kb = (lane & 16) ? 16 : 0;
  const _Float16* p = base + m * stride + k0 + kb;
  v8h lo = *(const v8h*)p;
  v8h hi = *(const v8h*)(p + 8);
  v16h a;
#pragma unroll
  for (int i = 0; i < 8; ++i) { a[i] = lo[i]; a[i + 8] = hi[i]; }
  return a;
}

// B fragment from K-major transposed weights WT[n][kp].
__device__ __forceinline__ v16h load_b_glb(const _Float16* WT, int kp, int n0,
                                           int k0, int lane) {
  const int n  = n0 + (lane & 15);
  const int ks = k0 + ((lane & 16) ? 16 : 0);
  return *(const v16h*)(WT + (size_t)n * kp + ks);
}

__device__ __forceinline__ void gemm_acc(const _Float16* sA, int kp,
                                         const _Float16* WT, int lane, int wv,
                                         v8f acc[4]) {
  const int n0 = wv * 64;
  for (int k0 = 0; k0 < kp; k0 += 32) {
    v16h af = load_a_lds(sA, kp, lane, k0);
#pragma unroll
    for (int t = 0; t < 4; ++t) {
      v16h bf = load_b_glb(WT, kp, n0 + t * 16, k0, lane);
      acc[t] = wmma16(af, bf, acc[t]);
    }
  }
}

__device__ __forceinline__ void store_relu_lds(_Float16* sO, const v8f acc[4],
                                               const float* bias, int lane,
                                               int wv) {
  const int rb   = (lane & 16) ? 8 : 0;
  const int ncol = lane & 15;
  const int n0   = wv * 64;
#pragma unroll
  for (int t = 0; t < 4; ++t) {
    const int n = n0 + t * 16 + ncol;
    const float b = bias[n];
#pragma unroll
    for (int r = 0; r < 8; ++r) {
      float v = acc[t][r] + b;
      sO[(r + rb) * DD + n] = (_Float16)(v > 0.f ? v : 0.f);
    }
  }
}

// ---------------- order-preserving float<->uint for atomic max -------------

__device__ __forceinline__ unsigned encf(float f) {
  unsigned u = __float_as_uint(f);
  return ((int)u >= 0) ? (u | 0x80000000u) : ~u;
}
__device__ __forceinline__ float decf(unsigned u) {
  unsigned v = (u & 0x80000000u) ? (u & 0x7FFFFFFFu) : ~u;
  return __uint_as_float(v);
}

// ---------------- utility / conversion kernels -----------------------------

__global__ __launch_bounds__(256) void k_fill_f32(float* p, float v, int n) {
  int i = blockIdx.x * 256 + threadIdx.x;
  if (i < n) p[i] = v;
}
__global__ __launch_bounds__(256) void k_fill_u32(unsigned* p, unsigned v, int n) {
  int i = blockIdx.x * 256 + threadIdx.x;
  if (i < n) p[i] = v;
}
__global__ __launch_bounds__(256) void k_conv16(const float* __restrict__ s,
                                                _Float16* __restrict__ d, int n) {
  int i = blockIdx.x * 256 + threadIdx.x;
  if (i < n) d[i] = (_Float16)s[i];
}
// f32 [rows x cols] -> f16 [rows x stride], pad cols..stride-1 with zeros
__global__ __launch_bounds__(256) void k_conv16_pad(const float* __restrict__ s,
                                                    _Float16* __restrict__ d,
                                                    int rows, int cols,
                                                    int stride) {
  int i = blockIdx.x * 256 + threadIdx.x;
  if (i >= rows * stride) return;
  int r = i / stride, c = i - r * stride;
  d[i] = (c < cols) ? (_Float16)s[(size_t)r * cols + c] : (_Float16)0.f;
}

// W [K x Nout] f32 -> WT [NP x KP] f16 K-major, identity K map, zero padded.
__global__ __launch_bounds__(256) void k_convT(const float* __restrict__ W,
                                               _Float16* __restrict__ WT,
                                               int K, int Nout, int KP, int NP) {
  const int total = NP * KP;
  for (int idx = blockIdx.x * blockDim.x + threadIdx.x; idx < total;
       idx += gridDim.x * blockDim.x) {
    const int n = idx / KP, k = idx % KP;
    float v = (k < K && n < Nout) ? W[(size_t)k * Nout + n] : 0.f;
    WT[idx] = (_Float16)v;
  }
}

// Segment-remapped transpose: padded-K position pstart[s]..+slen[s] maps to
// source rows sstart[s]..; everything else is a zero row.
struct SegMap { int nseg; int pstart[5]; int slen[5]; int sstart[5]; };

__global__ __launch_bounds__(256) void k_convT_seg(const float* __restrict__ W,
                                                   _Float16* __restrict__ WT,
                                                   int Nout, int KP, int NP,
                                                   SegMap m) {
  const int total = NP * KP;
  for (int idx = blockIdx.x * blockDim.x + threadIdx.x; idx < total;
       idx += gridDim.x * blockDim.x) {
    const int n = idx / KP, k = idx % KP;
    float v = 0.f;
    for (int s = 0; s < m.nseg; ++s) {
      if (k >= m.pstart[s] && k < m.pstart[s] + m.slen[s])
        v = W[(size_t)(m.sstart[s] + (k - m.pstart[s])) * Nout + n];
    }
    WT[idx] = (_Float16)v;
  }
}

// ---------------- edge MLPs + attention logits -----------------------------

__global__ __launch_bounds__(256) void k_edge(
    const _Float16* __restrict__ n_f16, const _Float16* __restrict__ w2v16,
    const _Float16* __restrict__ s_f16, const int* __restrict__ src,
    const int* __restrict__ dst,
    const _Float16* __restrict__ WeT, const float* __restrict__ b_e,
    const _Float16* __restrict__ WelT, const float* __restrict__ b_el,
    const float* __restrict__ W_a, const float* __restrict__ b_a,
    const float* __restrict__ W_al, const float* __restrict__ b_al,
    _Float16* __restrict__ e_f, float* __restrict__ aE, float* __restrict__ aL,
    unsigned* __restrict__ mmax, unsigned* __restrict__ mmaxl) {
  __shared__ _Float16 sA[16 * KP_E];   // 33792 B
  __shared__ _Float16 sO[16 * DD];     // 16384 B
  __shared__ float    sRed[256];

  const int tid  = threadIdx.x;
  const int lane = tid & 31;
  const int wv   = tid >> 5;
  const int e0   = blockIdx.x * 16;

  // ---- phase 1 gather: async f16 copies, 130 x 16B chunks per edge row ----
  for (int idx = tid; idx < 16 * 130; idx += 256) {
    const int row = idx / 130, c = idx - row * 130;
    const int e = e0 + row;
    const _Float16* g; int dh;
    if (c < 64)      { g = n_f16 + (size_t)src[e] * DD + (c << 3); dh = c << 3; }
    else if (c < 66) { g = s_f16 + (size_t)e * DSS + ((c - 64) << 3);
                       dh = DD + ((c - 64) << 3); }
    else             { g = n_f16 + (size_t)dst[e] * DD + ((c - 66) << 3);
                       dh = DD + DSS + ((c - 66) << 3); }
    async_cp16(&sA[row * KP_E + dh], g);
  }
  sA[(tid >> 4) * KP_E + 1040 + (tid & 15)] = (_Float16)0.f;  // K pad
  async_wait0();
  __syncthreads();
  {
    v8f acc[4] = {};
    gemm_acc(sA, KP_E, WeT, lane, wv, acc);
    store_relu_lds(sO, acc, b_e, lane, wv);
  }
  __syncthreads();
  for (int idx = tid; idx < 16 * DD / 8; idx += 256) {
    const int row = idx >> 6;
    const int c8  = (idx & 63) << 3;
    *(v8h*)(e_f + (size_t)(e0 + row) * DD + c8) = *(const v8h*)(sO + row * DD + c8);
  }
  {
    const int er = tid >> 4, c0 = tid & 15;
    float part = 0.f;
    for (int j = 0; j < 32; ++j) {
      const int c = c0 + (j << 4);
      part += (float)sO[er * DD + c] * W_a[c];
    }
    sRed[tid] = part;
  }
  __syncthreads();
  if (tid < 16) {
    float s = b_a[0];
    for (int c = 0; c < 16; ++c) s += sRed[tid * 16 + c];
    s = s > 0.f ? s : 0.2f * s;
    aE[e0 + tid] = s;
    atomicMax(&mmax[dst[e0 + tid]], encf(s));
  }
  __syncthreads();

  // ---- phase 2 gather: [src 304 | dst 304] halves, 76 chunks per row ----
  for (int idx = tid; idx < 16 * 76; idx += 256) {
    const int row = idx / 76, c = idx - row * 76;
    const int e = e0 + row;
    const _Float16* g; int dh;
    if (c < 38) { g = w2v16 + (size_t)src[e] * DWP + (c << 3); dh = c << 3; }
    else        { g = w2v16 + (size_t)dst[e] * DWP + ((c - 38) << 3);
                  dh = DWP + ((c - 38) << 3); }
    async_cp16(&sA[row * KP_L + dh], g);
  }
  async_wait0();
  __syncthreads();
  {
    v8f acc[4] = {};
    gemm_acc(sA, KP_L, WelT, lane, wv, acc);
    store_relu_lds(sO, acc, b_el, lane, wv);
  }
  __syncthreads();
  {
    const int er = tid >> 4, c0 = tid & 15;
    float part = 0.f;
    for (int j = 0; j < 32; ++j) {
      const int c = c0 + (j << 4);
      part += (float)sO[er * DD + c] * W_al[c];
    }
    sRed[tid] = part;
  }
  __syncthreads();
  if (tid < 16) {
    float s = b_al[0];
    for (int c = 0; c < 16; ++c) s += sRed[tid * 16 + c];
    s = s > 0.f ? s : 0.2f * s;
    aL[e0 + tid] = s;
    atomicMax(&mmaxl[dst[e0 + tid]], encf(s));
  }
}

// ---------------- softmax exp + segment sums -------------------------------

__global__ __launch_bounds__(256) void k_exp(
    const int* __restrict__ dst, const unsigned* __restrict__ mmax,
    const unsigned* __restrict__ mmaxl, float* __restrict__ aE,
    float* __restrict__ aL, float* __restrict__ ssum, float* __restrict__ ssuml) {
  const int e = blockIdx.x * 256 + threadIdx.x;
  if (e >= EE) return;
  const int d = dst[e];
  float v = expf(aE[e] - decf(mmax[d]));
  aE[e] = v;
  atomicAdd(&ssum[d], v);
  float v2 = expf(aL[e] - decf(mmaxl[d]));
  aL[e] = v2;
  atomicAdd(&ssuml[d], v2);
}

// ---------------- attention-weighted scatter aggregation -------------------

__global__ __launch_bounds__(256) void k_aggr(
    const int* __restrict__ src, const int* __restrict__ dst,
    const float* __restrict__ ssum, const float* __restrict__ ssuml,
    const float* __restrict__ aE, const float* __restrict__ aL,
    const _Float16* __restrict__ e_f, const float* __restrict__ w2v,
    float* __restrict__ z, float* __restrict__ z_l) {
  const int e   = blockIdx.x;
  const int tid = threadIdx.x;
  const int d = dst[e], s = src[e];
  const float alpha  = aE[e] / (ssum[d] + 1e-9f);
  const float alphal = aL[e] / (ssuml[d] + 1e-9f);
  atomicAdd(&z[(size_t)d * DD + tid], alpha * (float)e_f[(size_t)e * DD + tid]);
  atomicAdd(&z[(size_t)d * DD + tid + 256],
            alpha * (float)e_f[(size_t)e * DD + tid + 256]);
  if (tid < DWW)
    atomicAdd(&z_l[(size_t)d * DWP + tid], alphal * w2v[(size_t)s * DWW + tid]);
  if (tid + 256 < DWW)
    atomicAdd(&z_l[(size_t)d * DWP + tid + 256],
              alphal * w2v[(size_t)s * DWW + tid + 256]);
}

// ---------------- node MLPs ------------------------------------------------

__global__ __launch_bounds__(256) void k_node(
    const float* __restrict__ n_f, const float* __restrict__ z,
    const _Float16* __restrict__ WnT, const float* __restrict__ b_n,
    _Float16* __restrict__ n_new) {
  __shared__ _Float16 sA[16 * KP_N];
  const int tid  = threadIdx.x;
  const int lane = tid & 31;
  const int wv   = tid >> 5;
  const int nb   = blockIdx.x * 16;
  for (int i = 0; i < 16; ++i) {
    const int node = nb + i;
    for (int k = tid; k < KP_N; k += 256) {
      float v = (k < DD) ? n_f[(size_t)node * DD + k]
                         : z[(size_t)node * DD + (k - DD)];
      sA[i * KP_N + k] = (_Float16)v;
    }
  }
  __syncthreads();
  v8f acc[4] = {};
  gemm_acc(sA, KP_N, WnT, lane, wv, acc);
  const int rb = (lane & 16) ? 8 : 0;
  const int ncol = lane & 15;
  const int n0 = wv * 64;
#pragma unroll
  for (int t = 0; t < 4; ++t) {
    const int n = n0 + t * 16 + ncol;
    const float b = b_n[n];
#pragma unroll
    for (int r = 0; r < 8; ++r) {
      float v = acc[t][r] + b;
      n_new[(size_t)(nb + r + rb) * DD + n] = (_Float16)(v > 0.f ? v : 0.f);
    }
  }
}

__global__ __launch_bounds__(256) void k_nodelang(
    const float* __restrict__ w2v, const float* __restrict__ z_l,
    const _Float16* __restrict__ WnlT, const float* __restrict__ b_nl,
    _Float16* __restrict__ w_new) {
  __shared__ _Float16 sA[16 * KP_NL];
  const int tid  = threadIdx.x;
  const int lane = tid & 31;
  const int wv   = tid >> 5;
  const int nb   = blockIdx.x * 16;
  for (int i = 0; i < 16; ++i) {
    const int node = nb + i;
    for (int k = tid; k < KP_NL; k += 256) {
      float v;
      if (k < DWW)          v = w2v[(size_t)node * DWW + k];
      else if (k < 2 * DWW) v = z_l[(size_t)node * DWP + (k - DWW)];
      else                  v = 0.f;
      sA[i * KP_NL + k] = (_Float16)v;
    }
  }
  __syncthreads();
  const int rb = (lane & 16) ? 8 : 0;
  const int ncol = lane & 15;
  for (int t = wv; t < DWP / 16; t += 8) {
    v8f acc = {};
    const int n0 = t * 16;
    for (int k0 = 0; k0 < KP_NL; k0 += 32) {
      v16h af = load_a_lds(sA, KP_NL, lane, k0);
      v16h bf = load_b_glb(WnlT, KP_NL, n0, k0, lane);
      acc = wmma16(af, bf, acc);
    }
    const int n = n0 + ncol;
    const float b = (n < DWW) ? b_nl[n] : 0.f;
#pragma unroll
    for (int r = 0; r < 8; ++r) {
      float v = acc[r] + b;
      w_new[(size_t)(nb + r + rb) * DWP + n] = (_Float16)(v > 0.f ? v : 0.f);
    }
  }
}

// ---------------- edge readout ---------------------------------------------
// padded layout: [nsrc 512 | wsrc 304 | s_f 16 | wdst 304 | ndst 512 | pad 16]

__global__ __launch_bounds__(256) void k_readout(
    const _Float16* __restrict__ n_new, const _Float16* __restrict__ w_new,
    const _Float16* __restrict__ s_f16, const int* __restrict__ src,
    const int* __restrict__ dst, const _Float16* __restrict__ Wr1T,
    const float* __restrict__ b_r1, const float* __restrict__ W_r2,
    const float* __restrict__ b_r2, float* __restrict__ out) {
  __shared__ _Float16 sA[16 * KP_R];  // 53248 B (reused for h)
  const int tid  = threadIdx.x;
  const int lane = tid & 31;
  const int wv   = tid >> 5;
  const int e0   = blockIdx.x * 16;
  for (int idx = tid; idx < 16 * 206; idx += 256) {
    const int row = idx / 206, c = idx - row * 206;
    const int e = e0 + row;
    const _Float16* g; int dh;
    if (c < 64)       { g = n_new + (size_t)src[e] * DD + (c << 3); dh = c << 3; }
    else if (c < 102) { g = w_new + (size_t)src[e] * DWP + ((c - 64) << 3);
                        dh = 512 + ((c - 64) << 3); }
    else if (c < 104) { g = s_f16 + (size_t)e * DSS + ((c - 102) << 3);
                        dh = 816 + ((c - 102) << 3); }
    else if (c < 142) { g = w_new + (size_t)dst[e] * DWP + ((c - 104) << 3);
                        dh = 832 + ((c - 104) << 3); }
    else              { g = n_new + (size_t)dst[e] * DD + ((c - 142) << 3);
                        dh = 1136 + ((c - 142) << 3); }
    async_cp16(&sA[row * KP_R + dh], g);
  }
  sA[(tid >> 4) * KP_R + 1648 + (tid & 15)] = (_Float16)0.f;  // K pad
  async_wait0();
  __syncthreads();
  v8f acc[4] = {};
  gemm_acc(sA, KP_R, Wr1T, lane, wv, acc);
  __syncthreads();
  store_relu_lds(sA, acc, b_r1, lane, wv);  // h -> first 16*512 halves
  __syncthreads();
  if (tid < 16 * NC) {
    const int er = tid / NC, c = tid % NC;
    float sum = b_r2[c];
    for (int k = 0; k < DD; ++k)
      sum += (float)sA[er * DD + k] * W_r2[k * NC + c];
    out[(size_t)(e0 + er) * NC + c] = sum;
  }
}

// ---------------- host launcher --------------------------------------------

extern "C" void kernel_launch(void* const* d_in, const int* in_sizes, int n_in,
                              void* d_out, int out_size, void* d_ws,
                              size_t ws_size, hipStream_t stream) {
  const float* n_f  = (const float*)d_in[0];
  const float* w2v  = (const float*)d_in[1];
  const float* s_f  = (const float*)d_in[2];
  const int*   src  = (const int*)d_in[3];
  const int*   dst  = (const int*)d_in[4];
  const float* W_e  = (const float*)d_in[5];
  const float* b_e  = (const float*)d_in[6];
  const float* W_el = (const float*)d_in[7];
  const float* b_el = (const float*)d_in[8];
  const float* W_a  = (const float*)d_in[9];
  const float* b_a  = (const float*)d_in[10];
  const float* W_al = (const float*)d_in[11];
  const float* b_al = (const float*)d_in[12];
  const float* W_n  = (const float*)d_in[13];
  const float* b_n  = (const float*)d_in[14];
  const float* W_nl = (const float*)d_in[15];
  const float* b_nl = (const float*)d_in[16];
  const float* W_r1 = (const float*)d_in[17];
  const float* b_r1 = (const float*)d_in[18];
  const float* W_r2 = (const float*)d_in[19];
  const float* b_r2 = (const float*)d_in[20];
  float* out = (float*)d_out;

  char* wsb = (char*)d_ws;
  size_t off = 0;
  auto take = [&](size_t bytes) -> void* {
    void* p = wsb + off;
    off = (off + bytes + 255) & ~(size_t)255;
    return p;
  };

  _Float16* WeT   = (_Float16*)take((size_t)512 * KP_E * 2);
  _Float16* WelT  = (_Float16*)take((size_t)512 * KP_L * 2);
  _Float16* WnT   = (_Float16*)take((size_t)512 * KP_N * 2);
  _Float16* WnlT  = (_Float16*)take((size_t)DWP * KP_NL * 2);
  _Float16* Wr1T  = (_Float16*)take((size_t)512 * KP_R * 2);
  _Float16* n_f16 = (_Float16*)take((size_t)NN * DD * 2);
  _Float16* w2v16 = (_Float16*)take((size_t)NN * DWP * 2);
  _Float16* s_f16 = (_Float16*)take((size_t)EE * DSS * 2);
  _Float16* e_fB  = (_Float16*)take((size_t)EE * DD * 2);
  float*    aE    = (float*)take((size_t)EE * 4);
  float*    aL    = (float*)take((size_t)EE * 4);
  unsigned* mmax  = (unsigned*)take((size_t)NN * 4);
  unsigned* mmaxl = (unsigned*)take((size_t)NN * 4);
  float*    ssum  = (float*)take((size_t)NN * 4);
  float*    ssuml = (float*)take((size_t)NN * 4);
  float*    z     = (float*)take((size_t)NN * DD * 4);
  float*    z_l   = (float*)take((size_t)NN * DWP * 4);
  _Float16* n_new = (_Float16*)take((size_t)NN * DD * 2);
  _Float16* w_new = (_Float16*)take((size_t)NN * DWP * 2);

  // weights -> f16, transposed, K padded
  k_convT<<<256, 256, 0, stream>>>(W_e, WeT, 1040, 512, KP_E, 512);
  k_convT<<<256, 256, 0, stream>>>(W_n, WnT, 1024, 512, KP_N, 512);
  k_convT<<<256, 256, 0, stream>>>(W_nl, WnlT, 600, 300, KP_NL, DWP);
  {  // W_el remapped to [src 304 | dst 304]
    SegMap m; m.nseg = 2;
    m.pstart[0] = 0;   m.slen[0] = 300; m.sstart[0] = 0;
    m.pstart[1] = 304; m.slen[1] = 300; m.sstart[1] = 300;
    k_convT_seg<<<256, 256, 0, stream>>>(W_el, WelT, 512, KP_L, 512, m);
  }
  {  // W_r1 remapped to padded readout layout
    SegMap m; m.nseg = 5;
    m.pstart[0] = 0;    m.slen[0] = 512; m.sstart[0] = 0;
    m.pstart[1] = 512;  m.slen[1] = 300; m.sstart[1] = 512;
    m.pstart[2] = 816;  m.slen[2] = 16;  m.sstart[2] = 812;
    m.pstart[3] = 832;  m.slen[3] = 300; m.sstart[3] = 828;
    m.pstart[4] = 1136; m.slen[4] = 512; m.sstart[4] = 1128;
    k_convT_seg<<<256, 256, 0, stream>>>(W_r1, Wr1T, 512, KP_R, 512, m);
  }

  // activations -> f16 (enables async b128 global->LDS gathers)
  k_conv16<<<(NN * DD + 255) / 256, 256, 0, stream>>>(n_f, n_f16, NN * DD);
  k_conv16_pad<<<(NN * DWP + 255) / 256, 256, 0, stream>>>(w2v, w2v16, NN, DWW,
                                                           DWP);
  k_conv16<<<(EE * DSS + 255) / 256, 256, 0, stream>>>(s_f, s_f16, EE * DSS);

  // init accumulators
  k_fill_f32<<<(NN * DD + 255) / 256, 256, 0, stream>>>(z, 0.f, NN * DD);
  k_fill_f32<<<(NN * DWP + 255) / 256, 256, 0, stream>>>(z_l, 0.f, NN * DWP);
  k_fill_f32<<<(NN + 255) / 256, 256, 0, stream>>>(ssum, 0.f, NN);
  k_fill_f32<<<(NN + 255) / 256, 256, 0, stream>>>(ssuml, 0.f, NN);
  k_fill_u32<<<(NN + 255) / 256, 256, 0, stream>>>(mmax, 0x00800000u, NN);
  k_fill_u32<<<(NN + 255) / 256, 256, 0, stream>>>(mmaxl, 0x00800000u, NN);

  // pipeline
  k_edge<<<EE / 16, 256, 0, stream>>>(n_f16, w2v16, s_f16, src, dst, WeT, b_e,
                                      WelT, b_el, W_a, b_a, W_al, b_al, e_fB,
                                      aE, aL, mmax, mmaxl);
  k_exp<<<(EE + 255) / 256, 256, 0, stream>>>(dst, mmax, mmaxl, aE, aL, ssum,
                                              ssuml);
  k_aggr<<<EE, 256, 0, stream>>>(src, dst, ssum, ssuml, aE, aL, e_fB, w2v, z,
                                 z_l);
  k_node<<<NN / 16, 256, 0, stream>>>(n_f, z, WnT, b_n, n_new);
  k_nodelang<<<NN / 16, 256, 0, stream>>>(w2v, z_l, WnlT, b_nl, w_new);
  k_readout<<<EE / 16, 256, 0, stream>>>(n_new, w_new, s_f16, src, dst, Wr1T,
                                         b_r1, W_r2, b_r2, out);
}
